// Encoder_22539988369488
// MI455X (gfx1250) — matrix operand order repaired and checked by reading, not compiled, not run on hardware
//
#include <hip/hip_runtime.h>
#include <hip/hip_bf16.h>
#include <math.h>

// ---------------------------------------------------------------------------
// MI455X / gfx1250 implementation.
//  * all GEMMs via v_wmma_f32_16x16x32_bf16 (bf16 in, f32 accumulate)
//  * fast path: 128x64 block tiles, bf16 staged in LDS (ds_load_b128 frags),
//    one A-frag reused across 4 WMMAs per wave per K-chunk
//  * fused upsample+logits+softmax+zq kernel: outputs are ~402MB; that stage
//    is HBM-write bound, WMMAs stay register-resident
// workspace requirement: ~32 MB fp32 scratch in d_ws.
// ---------------------------------------------------------------------------

typedef __bf16 bf16_t;
typedef bf16_t v16bf __attribute__((ext_vector_type(16)));
typedef bf16_t v8bf  __attribute__((ext_vector_type(8)));
typedef float  v8f   __attribute__((ext_vector_type(8)));

#define B_SZ   2
#define NTOK   1024
#define DMOD   384
#define DBB    768
#define NHEAD  6
#define HD     64
#define MLPD   1536
#define EMB    32
#define NEMB   128
#define OPIX   512

// ISA 7.12.2: 16-bit A-matrix 16x32 K mapping for element e of the 16-elem frag
__device__ __forceinline__ int kmapA(int lane, int e) {
  return e + ((e >= 8) ? 8 : 0) + ((lane & 16) ? 8 : 0);
}
// 16-bit B-matrix: same K mapping as A (lane half selects 8-offset, e>=8 adds 8)
__device__ __forceinline__ int kmapB(int lane, int e) {
  return e + ((e >= 8) ? 8 : 0) + ((lane & 16) ? 8 : 0);
}
__device__ __forceinline__ int iclamp(int v, int lo, int hi) {
  return v < lo ? lo : (v > hi ? hi : v);
}
__device__ __forceinline__ v16bf cat16(v8bf lo, v8bf hi) {
  return __builtin_shufflevector(lo, hi, 0, 1, 2, 3, 4, 5, 6, 7,
                                 8, 9, 10, 11, 12, 13, 14, 15);
}
__device__ __forceinline__ v8bf pack8(float4 f0, float4 f1) {
  v8bf p;
  p[0] = (bf16_t)f0.x; p[1] = (bf16_t)f0.y; p[2] = (bf16_t)f0.z; p[3] = (bf16_t)f0.w;
  p[4] = (bf16_t)f1.x; p[5] = (bf16_t)f1.y; p[6] = (bf16_t)f1.z; p[7] = (bf16_t)f1.w;
  return p;
}

// ---------------------------------------------------------------------------
// Fast row-major GEMM: Cout[m,n] = act(sum_k A[m,k]W[n,k] + bias[n]) (+Cin)
// A:[M,K], W:[N,K], C:[M,N] row-major. Block tile 128x64, 8 waves, each wave
// computes a 16x64 strip (A frag reused over 4 WMMAs). bf16 panels in LDS.
// Requires M % 128 == 0; N ragged (clamped staging + guarded stores).
// ---------------------------------------------------------------------------
__global__ void __launch_bounds__(256)
k_wmma_gemm_fast(const float* __restrict__ A, const float* __restrict__ W,
                 const float* __restrict__ bias, const float* __restrict__ Cin,
                 float* __restrict__ Cout, int M, int N, int K, int act) {
  __shared__ bf16_t sA[128][32];      // 8 KB
  __shared__ bf16_t sW[64][32];       // 4 KB
  const int t = threadIdx.x, lane = t & 31, wv = t >> 5;
  const int bn = (N + 63) >> 6;
  const int m0 = (blockIdx.x / bn) * 128;
  const int n0 = (blockIdx.x % bn) * 64;

  v8f acc[4] = {{}, {}, {}, {}};

  for (int k0 = 0; k0 < K; k0 += 32) {
    // ---- stage A panel: 128x32 fp32 -> bf16, 8 elems/thread x 2 ----
#pragma unroll
    for (int j = 0; j < 2; ++j) {
      const int lin = (t + j * 256) * 8;
      const int row = lin >> 5, kk = lin & 31;
      const float* ap = &A[(long)(m0 + row) * K + k0 + kk];
      if (k0 + 32 < K) __builtin_prefetch(ap + 32, 0, 1);   // global_prefetch_b8
      const float4 f0 = *(const float4*)ap;
      const float4 f1 = *(const float4*)(ap + 4);
      *(v8bf*)&sA[row][kk] = pack8(f0, f1);
    }
    // ---- stage W panel: 64x32 fp32 -> bf16, 8 elems/thread ----
    {
      const int lin = t * 8;
      const int row = lin >> 5, kk = lin & 31;
      const int rg = (n0 + row < N) ? (n0 + row) : (N - 1);
      const float* wp = &W[(long)rg * K + k0 + kk];
      const float4 f0 = *(const float4*)wp;
      const float4 f1 = *(const float4*)(wp + 4);
      *(v8bf*)&sW[row][kk] = pack8(f0, f1);
    }
    __syncthreads();

    // ---- A fragment: two contiguous 8-elem K-runs (ds_load_b128 x2) ----
    const int arow = wv * 16 + (lane & 15);
    const int aoff = (lane & 16) ? 8 : 0;
    const v16bf a = cat16(*(const v8bf*)&sA[arow][aoff],
                          *(const v8bf*)&sA[arow][aoff + 16]);
    const int boff = (lane & 16) ? 8 : 0;
#pragma unroll
    for (int tt = 0; tt < 4; ++tt) {
      const int brow = tt * 16 + (lane & 15);
      const v16bf b = cat16(*(const v8bf*)&sW[brow][boff],
                            *(const v8bf*)&sW[brow][boff + 16]);
      acc[tt] = __builtin_amdgcn_wmma_f32_16x16x32_bf16(false, a, false, b,
                                                        (short)0, acc[tt], false, false);
    }
    __syncthreads();
  }

  const int mBase = m0 + wv * 16 + ((lane & 16) ? 8 : 0);
#pragma unroll
  for (int tt = 0; tt < 4; ++tt) {
    const int n = n0 + tt * 16 + (lane & 15);
    if (n >= N) continue;
    const float bs = bias ? bias[n] : 0.0f;
#pragma unroll
    for (int r = 0; r < 8; ++r) {
      const long m = mBase + r;
      float v = acc[tt][r] + bs;
      if (act == 1) v = 0.5f * v * (1.0f + erff(v * 0.70710678118654752f));
      if (Cin) v += Cin[m * N + n];
      Cout[m * N + n] = v;
    }
  }
}

// ---------------------------------------------------------------------------
// Generic strided WMMA GEMM (for the small per-head ctx/att GEMMs whose K
// strides are non-unit). One 16x16 tile per wave.
// ---------------------------------------------------------------------------
__global__ void k_wmma_gemm(const float* __restrict__ A, long As, long Aks,
                            const float* __restrict__ W, long Ws, long Wks,
                            float* __restrict__ Cout, long Cos,
                            int M, int N, int K) {
  const int lane = threadIdx.x & 31;
  const int wv   = threadIdx.x >> 5;
  const int tilesN = N >> 4;
  const int tilesM = M >> 4;
  const int tile = blockIdx.x * (blockDim.x >> 5) + wv;
  if (tile >= tilesM * tilesN) return;          // wave-uniform exit (EXEC stays full)
  const int tm = tile / tilesN, tn = tile % tilesN;
  const long rA = tm * 16 + (lane & 15);
  const long rW = tn * 16 + (lane & 15);

  v8f acc = {};
  for (int k0 = 0; k0 < K; k0 += 32) {
    if (k0 + 32 < K)
      __builtin_prefetch(&A[rA * As + (long)(k0 + 32) * Aks], 0, 1);
    v16bf a, b;
#pragma unroll
    for (int e = 0; e < 16; ++e) {
      const long ka = k0 + kmapA(lane, e);
      const long kb = k0 + kmapB(lane, e);
      a[e] = (bf16_t)A[rA * As + ka * Aks];
      b[e] = (bf16_t)W[rW * Ws + kb * Wks];
    }
    acc = __builtin_amdgcn_wmma_f32_16x16x32_bf16(false, a, false, b,
                                                  (short)0, acc, false, false);
  }

  const int mBase = tm * 16 + ((lane & 16) ? 8 : 0);
  const int n     = tn * 16 + (lane & 15);
#pragma unroll
  for (int r = 0; r < 8; ++r)
    Cout[(long)(mBase + r) * Cos + n] = acc[r];
}

// ---------------------------------------------------------------------------
// Row LayerNorm, D features per row. One block per row.
// ---------------------------------------------------------------------------
__global__ void k_layernorm(const float* __restrict__ x, const float* __restrict__ g,
                            const float* __restrict__ b, float* __restrict__ y, int D) {
  __shared__ float red[128];
  const int row = blockIdx.x, t = threadIdx.x;
  const float* xr = x + (long)row * D;
  float s = 0.0f;
  for (int i = t; i < D; i += 128) s += xr[i];
  red[t] = s; __syncthreads();
  for (int st = 64; st > 0; st >>= 1) { if (t < st) red[t] += red[t + st]; __syncthreads(); }
  const float mu = red[0] / D; __syncthreads();
  float v = 0.0f;
  for (int i = t; i < D; i += 128) { const float d = xr[i] - mu; v += d * d; }
  red[t] = v; __syncthreads();
  for (int st = 64; st > 0; st >>= 1) { if (t < st) red[t] += red[t + st]; __syncthreads(); }
  const float rstd = rsqrtf(red[0] / D + 1e-5f);
  for (int i = t; i < D; i += 128)
    y[(long)row * D + i] = (xr[i] - mu) * rstd * g[i] + b[i];
}

// softmax over feature dim (64) per (b,n,h), then *HD^-0.5; in-place on q slice
__global__ void k_softmax_q(float* __restrict__ qkv) {
  __shared__ float red[64];
  const int idx = blockIdx.x;
  const int h = idx % NHEAD, bn = idx / NHEAD;
  float* row = qkv + (long)bn * (3 * DMOD) + h * HD;
  const int t = threadIdx.x;
  const float v = row[t];
  red[t] = v; __syncthreads();
  for (int st = 32; st > 0; st >>= 1) { if (t < st) red[t] = fmaxf(red[t], red[t + st]); __syncthreads(); }
  const float mx = red[0]; __syncthreads();
  const float e = __expf(v - mx);
  red[t] = e; __syncthreads();
  for (int st = 32; st > 0; st >>= 1) { if (t < st) red[t] += red[t + st]; __syncthreads(); }
  row[t] = e / red[0] * 0.125f;             // HD^-0.5 = 1/8
}

// softmax over sequence dim (1024) per (b,h,d); in-place on k slice
__global__ void k_softmax_k(float* __restrict__ qkv) {
  __shared__ float red[256];
  const int idx = blockIdx.x;
  const int d = idx % HD, bh = idx / HD;
  const int h = bh % NHEAD, b = bh / NHEAD;
  float* base = qkv + (long)b * NTOK * (3 * DMOD) + DMOD + h * HD + d;
  const int t = threadIdx.x;
  float loc[4], mx = -3.4e38f;
#pragma unroll
  for (int j = 0; j < 4; ++j) { loc[j] = base[(long)(t + j * 256) * (3 * DMOD)]; mx = fmaxf(mx, loc[j]); }
  red[t] = mx; __syncthreads();
  for (int st = 128; st > 0; st >>= 1) { if (t < st) red[t] = fmaxf(red[t], red[t + st]); __syncthreads(); }
  mx = red[0]; __syncthreads();
  float s = 0.0f;
#pragma unroll
  for (int j = 0; j < 4; ++j) { loc[j] = __expf(loc[j] - mx); s += loc[j]; }
  red[t] = s; __syncthreads();
  for (int st = 128; st > 0; st >>= 1) { if (t < st) red[t] += red[t + st]; __syncthreads(); }
  const float inv = 1.0f / red[0];
#pragma unroll
  for (int j = 0; j < 4; ++j) base[(long)(t + j * 256) * (3 * DMOD)] = loc[j] * inv;
}

// key_cb[l,d] = sum_o Kw[d,o]E[l,o] ; val_cb[l,d] = sum_o Vw[d,o]E[l,o]
__global__ void k_codebook(const float* __restrict__ Kw, const float* __restrict__ Vw,
                           const float* __restrict__ E, float* __restrict__ kcb,
                           float* __restrict__ vcb) {
  const int i = blockIdx.x * blockDim.x + threadIdx.x;
  if (i >= 2 * NEMB * EMB) return;
  const int which = i >> 12, ld = i & 4095, l = ld >> 5, d = ld & 31;
  const float* Wm = which ? Vw : Kw;
  float s = 0.0f;
  for (int o = 0; o < 512; ++o) s += Wm[d * 512 + o] * E[l * 512 + o];
  (which ? vcb : kcb)[l * EMB + d] = s;
}

// ---------------------------------------------------------------------------
// Fused: bilinear x16 upsample -> z_e_x, logits = z_e . key_cb, softmax(128),
// z_after = P . val_cb. One block per (b, output row); each wave owns 16 px.
// ---------------------------------------------------------------------------
__global__ void k_vq_fused(const float* __restrict__ y, const float* __restrict__ kcb,
                           const float* __restrict__ vcb, float* __restrict__ out) {
  const long PLANE = (long)OPIX * OPIX;
  const long O1 = (long)B_SZ * EMB * PLANE;         // logits offset
  const long O2 = O1 + (long)B_SZ * NEMB * PLANE;   // z_after offset

  __shared__ float  lds_y[2][32][32];               // two source rows, 8KB
  __shared__ bf16_t lds_p[8][16 * 128];             // per-wave prob transpose, 32KB

  const int b  = blockIdx.x >> 9;
  const int oy = blockIdx.x & 511;
  const int tid = threadIdx.x, lane = tid & 31, wv = tid >> 5;
  const int half8 = (lane & 16) ? 8 : 0;

  // half-pixel bilinear source rows (clamped)
  const float fy  = (oy + 0.5f) * 0.0625f - 0.5f;
  const float fy0 = floorf(fy);
  const float wy1 = fy - fy0, wy0 = 1.0f - wy1;
  const int iy0 = iclamp((int)fy0, 0, 31);
  const int iy1 = iclamp((int)fy0 + 1, 0, 31);

  for (int i = tid; i < 2048; i += 256) {
    const int ry = i >> 10, rem = i & 1023, sx = rem >> 5, ch = rem & 31;
    const int ty = ry ? iy1 : iy0;
    lds_y[ry][sx][ch] = y[((long)b * NTOK + ty * 32 + sx) * EMB + ch];
  }
  __syncthreads();

  // resident codebook B-fragments (bf16)
  v16bf kf[8];
#pragma unroll
  for (int t = 0; t < 8; ++t) {
    const int n = t * 16 + (lane & 15);
#pragma unroll
    for (int e = 0; e < 16; ++e) kf[t][e] = (bf16_t)kcb[n * EMB + kmapB(lane, e)];
  }
  v16bf vf[2][4];
#pragma unroll
  for (int te = 0; te < 2; ++te) {
    const int eo = te * 16 + (lane & 15);
#pragma unroll
    for (int c = 0; c < 4; ++c)
#pragma unroll
      for (int e = 0; e < 16; ++e)
        vf[te][c][e] = (bf16_t)vcb[(c * 32 + kmapB(lane, e)) * EMB + eo];
  }

  for (int it = 0; it < 4; ++it) {
    const int ox_base = it * 128 + wv * 16;
    const int ox = ox_base + (lane & 15);
    const float fx  = (ox + 0.5f) * 0.0625f - 0.5f;
    const float fx0 = floorf(fx);
    const float wx1 = fx - fx0, wx0 = 1.0f - wx1;
    const int ix0 = iclamp((int)fx0, 0, 31);
    const int ix1 = iclamp((int)fx0 + 1, 0, 31);

    // bilinear z_e for this lane's 16 channels -> write z_e_x + build A frag
    v16bf a;
#pragma unroll
    for (int e = 0; e < 16; ++e) {
      const int ch = kmapA(lane, e);
      const float v = wy0 * (wx0 * lds_y[0][ix0][ch] + wx1 * lds_y[0][ix1][ch]) +
                      wy1 * (wx0 * lds_y[1][ix0][ch] + wx1 * lds_y[1][ix1][ch]);
      out[((long)(b * EMB + ch) * OPIX + oy) * OPIX + ox] = v;
      a[e] = (bf16_t)v;
    }

    // logits: 8 WMMAs, [16px x 32ch] x [32ch x 16cls]
    v8f lg[8];
#pragma unroll
    for (int t = 0; t < 8; ++t) {
      v8f z = {};
      lg[t] = __builtin_amdgcn_wmma_f32_16x16x32_bf16(false, a, false, kf[t],
                                                      (short)0, z, false, false);
    }
#pragma unroll
    for (int t = 0; t < 8; ++t) {
      const int n = t * 16 + (lane & 15);
#pragma unroll
      for (int r = 0; r < 8; ++r)
        out[O1 + ((long)(b * NEMB + n) * OPIX + oy) * OPIX + (ox_base + half8 + r)] = lg[t][r];
    }

    // softmax over 128 classes per pixel: 8 tiles locally + 16-lane xor-shuffle
#pragma unroll
    for (int r = 0; r < 8; ++r) {
      float mx = -3.4e38f;
#pragma unroll
      for (int t = 0; t < 8; ++t) mx = fmaxf(mx, lg[t][r]);
      for (int msk = 1; msk < 16; msk <<= 1) mx = fmaxf(mx, __shfl_xor(mx, msk, 32));
      float s = 0.0f;
#pragma unroll
      for (int t = 0; t < 8; ++t) { const float e2 = __expf(lg[t][r] - mx); lg[t][r] = e2; s += e2; }
      for (int msk = 1; msk < 16; msk <<= 1) s += __shfl_xor(s, msk, 32);
      const float inv = 1.0f / s;
#pragma unroll
      for (int t = 0; t < 8; ++t) lg[t][r] *= inv;
    }

    // transpose probs (D layout -> A layout) through LDS
#pragma unroll
    for (int t = 0; t < 8; ++t) {
      const int n = t * 16 + (lane & 15);
#pragma unroll
      for (int r = 0; r < 8; ++r)
        lds_p[wv][(half8 + r) * 128 + n] = (bf16_t)lg[t][r];
    }
    __syncthreads();
    v16bf pa[4];
#pragma unroll
    for (int c = 0; c < 4; ++c)
#pragma unroll
      for (int e = 0; e < 16; ++e)
        pa[c][e] = lds_p[wv][(lane & 15) * 128 + c * 32 + kmapA(lane, e)];

    // z_after: 2 tiles x 4 K-chunks = 8 WMMAs
#pragma unroll
    for (int te = 0; te < 2; ++te) {
      v8f acc = {};
#pragma unroll
      for (int c = 0; c < 4; ++c)
        acc = __builtin_amdgcn_wmma_f32_16x16x32_bf16(false, pa[c], false, vf[te][c],
                                                      (short)0, acc, false, false);
      const int eo = te * 16 + (lane & 15);
#pragma unroll
      for (int r = 0; r < 8; ++r)
        out[O2 + ((long)(b * EMB + eo) * OPIX + oy) * OPIX + (ox_base + half8 + r)] = acc[r];
    }
    __syncthreads();
  }
}

// ---------------------------------------------------------------------------
extern "C" void kernel_launch(void* const* d_in, const int* in_sizes, int n_in,
                              void* d_out, int out_size, void* d_ws, size_t ws_size,
                              hipStream_t stream) {
  (void)in_sizes; (void)n_in; (void)out_size; (void)ws_size;
  const float* x     = (const float*)d_in[0];
  const float* Wpre  = (const float*)d_in[1];
  const float* bpre  = (const float*)d_in[2];
  const float* lnf_g = (const float*)d_in[25];
  const float* lnf_b = (const float*)d_in[26];
  const float* Whead = (const float*)d_in[27];
  const float* bhead = (const float*)d_in[28];
  const float* Emb   = (const float*)d_in[29];
  const float* Kw    = (const float*)d_in[30];
  const float* Vw    = (const float*)d_in[31];

  float* ws  = (float*)d_ws;
  float* OUT = ws;                    // 2048*384
  float* H   = OUT + 786432;          // 2048*384
  float* QKV = H + 786432;            // 2048*1152
  float* CTX = QKV + 2359296;         // 12*64*64
  float* ATT = CTX + 49152;           // 2048*384
  float* U   = ATT + 786432;          // 2048*1536
  float* Y   = U + 3145728;           // 2048*32
  float* KCB = Y + 65536;             // 128*32
  float* VCB = KCB + 4096;            // 128*32

  auto gemm_fast = [&](const float* A, const float* W, const float* bias,
                       const float* Cin, float* Cout, int M, int N, int K, int act) {
    const int blocks = (M / 128) * ((N + 63) / 64);
    k_wmma_gemm_fast<<<blocks, 256, 0, stream>>>(A, W, bias, Cin, Cout, M, N, K, act);
  };
  auto gemm_strided = [&](const float* A, long As, long Aks,
                          const float* W, long Ws, long Wks,
                          float* Cout, long Cos, int M, int N, int K) {
    const int tiles = (M / 16) * (N / 16);
    k_wmma_gemm<<<(tiles + 7) / 8, 256, 0, stream>>>(A, As, Aks, W, Ws, Wks,
                                                     Cout, Cos, M, N, K);
  };

  const int MTOT = B_SZ * NTOK;       // 2048 rows

  // pre-projection: OUT = x @ Wpre^T + bpre
  gemm_fast(x, Wpre, bpre, nullptr, OUT, MTOT, DMOD, DBB, 0);

  for (int l = 0; l < 2; ++l) {
    const int pb = 3 + l * 11;
    const float* ln1_g = (const float*)d_in[pb + 0];
    const float* ln1_b = (const float*)d_in[pb + 1];
    const float* Wqkv  = (const float*)d_in[pb + 2];
    const float* Wo    = (const float*)d_in[pb + 3];
    const float* bo    = (const float*)d_in[pb + 4];
    const float* ln2_g = (const float*)d_in[pb + 5];
    const float* ln2_b = (const float*)d_in[pb + 6];
    const float* W1    = (const float*)d_in[pb + 7];
    const float* b1    = (const float*)d_in[pb + 8];
    const float* W2    = (const float*)d_in[pb + 9];
    const float* b2    = (const float*)d_in[pb + 10];

    k_layernorm<<<MTOT, 128, 0, stream>>>(OUT, ln1_g, ln1_b, H, DMOD);
    gemm_fast(H, Wqkv, nullptr, nullptr, QKV, MTOT, 3 * DMOD, DMOD, 0);
    k_softmax_q<<<B_SZ * NTOK * NHEAD, 64, 0, stream>>>(QKV);
    k_softmax_k<<<B_SZ * NHEAD * HD, 256, 0, stream>>>(QKV);

    for (int bh = 0; bh < B_SZ * NHEAD; ++bh) {
      const int bb = bh / NHEAD, hh = bh % NHEAD;
      const float* kb = QKV + (long)bb * NTOK * (3 * DMOD) + DMOD + hh * HD;
      const float* vb = QKV + (long)bb * NTOK * (3 * DMOD) + 2 * DMOD + hh * HD;
      const float* qb = QKV + (long)bb * NTOK * (3 * DMOD) + hh * HD;
      float* cp = CTX + bh * HD * HD;
      float* ap = ATT + (long)bb * NTOK * DMOD + hh * HD;
      // ctx = k^T v : M=64(d), N=64(e), K=1024(n)
      gemm_strided(kb, 1, 3 * DMOD, vb, 1, 3 * DMOD, cp, HD, HD, HD, NTOK);
      // att = q @ ctx : M=1024(n), N=64(e), K=64(d); W[n=e,k=d] = ctx[d*64+e]
      gemm_strided(qb, 3 * DMOD, 1, cp, 1, HD, ap, DMOD, NTOK, HD, HD);
    }
    // OUT += ATT @ Wo^T + bo
    gemm_fast(ATT, Wo, bo, OUT, OUT, MTOT, DMOD, DMOD, 0);
    // MLP
    k_layernorm<<<MTOT, 128, 0, stream>>>(OUT, ln2_g, ln2_b, H, DMOD);
    gemm_fast(H, W1, b1, nullptr, U, MTOT, MLPD, DMOD, 1);
    gemm_fast(U, W2, b2, OUT, OUT, MTOT, DMOD, MLPD, 0);
  }

  // head
  k_layernorm<<<MTOT, 128, 0, stream>>>(OUT, lnf_g, lnf_b, H, DMOD);
  gemm_fast(H, Whead, bhead, nullptr, Y, MTOT, EMB, DMOD, 0);

  // codebooks + fused upsample/logits/softmax/z_after
  k_codebook<<<(2 * NEMB * EMB + 255) / 256, 256, 0, stream>>>(Kw, Vw, Emb, KCB, VCB);
  k_vq_fused<<<B_SZ * OPIX, 256, 0, stream>>>(Y, KCB, VCB, (float*)d_out);
}
